// GraphAttentionalLayer_1305670058149
// MI455X (gfx1250) — matrix operand order repaired and checked by reading, not compiled, not run on hardware
//
#include <hip/hip_runtime.h>
#include <hip/hip_bf16.h>

// ---------------------------------------------------------------------------
// GAT layer for MI455X (gfx1250, wave32, WMMA, async global->LDS).
//
// Softmax over m of (src[n]+tgt[m]) is invariant to the row-constant src[n]:
//   attn[h,n,m] = adj[n,m]*w[h,m] / sum_m adj[n,m]*w[h,m],  w = exp(tgt - max_h)
// => onum = adj_f16 @ P, P packs w*ht (1024 cols) + w (4 denom cols).
// Pipeline:
//   1) k_gemm1    : ht = h@W (WMMA f16, f32 accum)        [4096 x 1024] f32
//   2) k_tgt      : tgt[h,m] = leakyrelu(ht) . a_tgt      [4 x 4096] f32
//   3) k_max      : per-head max of tgt                   [4] f32
//   4) k_build_T  : BmatT f16 [1152 x 4096] = (w*ht)^T  (K-contiguous per col)
//      k_build_den: rows 1024..1027 = w, rows 1028..1151 = 0
//   5) k_gemm2    : onum = adj_f16 @ BmatT^T (WMMA; B tile staged with
//                   global_load_async_to_lds_b128)        [4096 x 1152] f32
//   6) k_final    : out[n,d] = 0.25 * sum_h onum[n,h*256+d]/onum[n,1024+h]
// ---------------------------------------------------------------------------

typedef _Float16 f16_t;
typedef __attribute__((ext_vector_type(8)))  _Float16 v8h;
typedef __attribute__((ext_vector_type(16))) _Float16 v16h;
typedef __attribute__((ext_vector_type(8)))  float    v8f;
typedef __attribute__((ext_vector_type(4)))  int      v4i;

#define N_NODES   4096
#define F_IN      512
#define HD        1024        // N_HEADS * N_HIDDEN
#define NH        4
#define DH        256
#define BCOLS     1152        // 1024 P cols + 4 denom cols, padded to 9*128
#define LDA       40          // LDS row stride in halves (80B = 5*16B: every
                              // b128 fragment/async access stays 16B aligned)

// ---- CDNA5 async global->LDS helpers (guarded; sync fallback) --------------
__device__ __forceinline__ void async_copy_b128(const f16_t* g, f16_t* l) {
#if defined(__gfx1250__) && __has_builtin(__builtin_amdgcn_global_load_async_to_lds_b128)
  __builtin_amdgcn_global_load_async_to_lds_b128(
      (__attribute__((address_space(1))) v4i*)g,
      (__attribute__((address_space(3))) v4i*)l, 0, 0);
#else
  *(v8h*)l = *(const v8h*)g;                 // sync fallback through VGPRs
#endif
}

__device__ __forceinline__ void wait_async() {
#if defined(__gfx1250__) && __has_builtin(__builtin_amdgcn_s_wait_asynccnt)
  __builtin_amdgcn_s_wait_asynccnt(0);
#elif defined(__gfx1250__)
  asm volatile("s_wait_asynccnt 0x0" ::: "memory");
#endif
}

// A fragment (16x32 f16, ISA layout): lane<16 holds row=lane, K={0..7,16..23};
// lane>=16 holds row=lane-16, K={8..15,24..31}. Two contiguous b128 reads.
__device__ __forceinline__ v16h load_frag_A(const f16_t* lds, int row, int lane) {
  const int koff = (lane < 16) ? 0 : 8;
  union { v16h v; v8h h[2]; } u;
  u.h[0] = *(const v8h*)(lds + row * LDA + koff);
  u.h[1] = *(const v8h*)(lds + row * LDA + 16 + koff);
  return u.v;
}

// B fragment (32x16 f16): lane<16 holds col=lane, K={0..7,16..23}; lane>=16 holds
// col=lane-16, K={8..15,24..31}. Blds is stored K-contiguous per column.
__device__ __forceinline__ v16h load_frag_B(const f16_t* lds, int col, int lane) {
  const int koff = (lane < 16) ? 0 : 8;
  union { v16h v; v8h h[2]; } u;
  u.h[0] = *(const v8h*)(lds + col * LDA + koff);
  u.h[1] = *(const v8h*)(lds + col * LDA + 16 + koff);
  return u.v;
}

// ------------------------- GEMM1: ht = h @ W --------------------------------
__global__ __launch_bounds__(256) void k_gemm1(const float* __restrict__ hmat,
                                               const float* __restrict__ Wmat,
                                               float* __restrict__ ht) {
  __shared__ f16_t Alds[128 * LDA];
  __shared__ f16_t Blds[128 * LDA];

  const int lane   = threadIdx.x & 31;
  const int wave   = threadIdx.x >> 5;
  const int waveM  = wave >> 2;                 // 0..1  (64 rows each)
  const int waveN  = wave & 3;                  // 0..3  (32 cols each)
  const int tile_m = blockIdx.y * 128;
  const int tile_n = blockIdx.x * 128;

  v8f acc[4][2];
#pragma unroll
  for (int mt = 0; mt < 4; ++mt)
#pragma unroll
    for (int nt = 0; nt < 2; ++nt) acc[mt][nt] = {};

  for (int kk = 0; kk < F_IN; kk += 32) {
    // Stage A tile: 128x32 f32 -> f16 LDS (row-major in K)
#pragma unroll
    for (int f = 0; f < 4; ++f) {
      const int linear = threadIdx.x + f * 256;          // 0..1023
      const int row = linear >> 3;
      const int c4  = (linear & 7) << 2;
      const float4 v = *(const float4*)(hmat + (size_t)(tile_m + row) * F_IN + kk + c4);
      f16_t* dst = &Alds[row * LDA + c4];
      dst[0] = (f16_t)v.x; dst[1] = (f16_t)v.y; dst[2] = (f16_t)v.z; dst[3] = (f16_t)v.w;
    }
    // Stage B tile: 32x128 f32 -> f16 LDS, transposed (K contiguous per col)
#pragma unroll
    for (int f = 0; f < 4; ++f) {
      const int linear = threadIdx.x + f * 256;          // 0..1023
      const int row = linear >> 5;                       // 0..31  (= k)
      const int c4  = (linear & 31) << 2;
      const float4 v = *(const float4*)(Wmat + (size_t)(kk + row) * HD + tile_n + c4);
      Blds[(c4 + 0) * LDA + row] = (f16_t)v.x;
      Blds[(c4 + 1) * LDA + row] = (f16_t)v.y;
      Blds[(c4 + 2) * LDA + row] = (f16_t)v.z;
      Blds[(c4 + 3) * LDA + row] = (f16_t)v.w;
    }
    if (kk + 32 < F_IN) {   // prefetch next K tile -> global_prefetch_b8
      __builtin_prefetch(hmat + (size_t)(tile_m + (threadIdx.x >> 1)) * F_IN + kk + 32, 0, 0);
      __builtin_prefetch(Wmat + (size_t)(kk + 32 + (threadIdx.x >> 3)) * HD + tile_n, 0, 0);
    }
    __syncthreads();

    v16h bfr[2];
#pragma unroll
    for (int nt = 0; nt < 2; ++nt)
      bfr[nt] = load_frag_B(Blds, waveN * 32 + nt * 16 + (lane & 15), lane);
#pragma unroll
    for (int mt = 0; mt < 4; ++mt) {
      const v16h afr = load_frag_A(Alds, waveM * 64 + mt * 16 + (lane & 15), lane);
#pragma unroll
      for (int nt = 0; nt < 2; ++nt)
        acc[mt][nt] = __builtin_amdgcn_wmma_f32_16x16x32_f16(
            false, afr, false, bfr[nt], (short)0, acc[mt][nt], false, false);
    }
    __syncthreads();
  }

  // C layout: VGPR i -> (M = i + (lane<16?0:8), N = lane&15)
#pragma unroll
  for (int mt = 0; mt < 4; ++mt)
#pragma unroll
    for (int nt = 0; nt < 2; ++nt) {
      const int col   = tile_n + waveN * 32 + nt * 16 + (lane & 15);
      const int rbase = tile_m + waveM * 64 + mt * 16 + ((lane < 16) ? 0 : 8);
#pragma unroll
      for (int i = 0; i < 8; ++i)
        ht[(size_t)(rbase + i) * HD + col] = acc[mt][nt][i];
    }
}

// ---------------- tgt[h,m] = leaky_relu(ht[m, h-block]) . a_tgt -------------
__global__ __launch_bounds__(256) void k_tgt(const float* __restrict__ ht,
                                             const float* __restrict__ a,
                                             float* __restrict__ tgt) {
  const int lane = threadIdx.x & 31;
  const int wave = threadIdx.x >> 5;
  const int hm   = blockIdx.x * 8 + wave;       // 0..16383
  const int hh   = hm >> 12;
  const int m    = hm & (N_NODES - 1);
  float s = 0.f;
  for (int j = lane; j < DH; j += 32) {
    float x = ht[(size_t)m * HD + hh * DH + j];
    x = (x > 0.f) ? x : 0.1f * x;
    s += x * a[hh * 2 * DH + DH + j];           // target half of a
  }
#pragma unroll
  for (int off = 16; off > 0; off >>= 1) s += __shfl_xor(s, off, 32);
  if (lane == 0) tgt[hh * N_NODES + m] = s;
}

// ----------------------- per-head max reduction -----------------------------
__global__ __launch_bounds__(256) void k_max(const float* __restrict__ tgt,
                                             float* __restrict__ hmax) {
  __shared__ float red[256];
  const int hh = blockIdx.x;
  float m = -1e30f;
  for (int i = threadIdx.x; i < N_NODES; i += 256)
    m = fmaxf(m, tgt[hh * N_NODES + i]);
  red[threadIdx.x] = m;
  __syncthreads();
  for (int s = 128; s > 0; s >>= 1) {
    if (threadIdx.x < s) red[threadIdx.x] = fmaxf(red[threadIdx.x], red[threadIdx.x + s]);
    __syncthreads();
  }
  if (threadIdx.x == 0) hmax[hh] = red[0];
}

// ---- BmatT[col][m] (col<1024): w[h,m]*ht[m,col] -- LDS-tiled transpose -----
// Block: 32 cols (within one head) x 256 m. Coalesced f32 reads along d,
// coalesced f16 writes along m.
__global__ __launch_bounds__(256) void k_build_T(const float* __restrict__ ht,
                                                 const float* __restrict__ tgt,
                                                 const float* __restrict__ hmax,
                                                 f16_t* __restrict__ BmatT) {
  __shared__ float wlds[256];
  __shared__ f16_t T[32][256 + 8];
  const int hh = blockIdx.z;
  const int c0 = blockIdx.x * 32;               // 0..224 within head
  const int m0 = blockIdx.y * 256;
  const int t  = threadIdx.x;

  wlds[t] = __expf(tgt[hh * N_NODES + m0 + t] - hmax[hh]);
  __syncthreads();

  const int cl = t & 31;                        // local col
  const int mg = t >> 5;                        // 0..7
#pragma unroll
  for (int i = 0; i < 32; ++i) {                // read ht coalesced over cols
    const int ml = i * 8 + mg;
    const float v = ht[(size_t)(m0 + ml) * HD + hh * DH + c0 + cl] * wlds[ml];
    T[cl][ml] = (f16_t)v;
  }
  __syncthreads();
#pragma unroll
  for (int i = 0; i < 4; ++i) {                 // write BmatT coalesced over m
    const int clw = mg + i * 8;
#pragma unroll
    for (int j = 0; j < 8; ++j) {
      const int ml = j * 32 + (t & 31);
      BmatT[(size_t)(hh * DH + c0 + clw) * N_NODES + m0 + ml] = T[clw][ml];
    }
  }
}

// ---- BmatT rows 1024..1027 = w[h,m]; rows 1028..1151 = 0 -------------------
__global__ __launch_bounds__(256) void k_build_den(const float* __restrict__ tgt,
                                                   const float* __restrict__ hmax,
                                                   f16_t* __restrict__ BmatT) {
  const int idx = blockIdx.x * 256 + threadIdx.x;       // < 128*4096
  const int r = idx >> 12;                              // 0..127
  const int m = idx & (N_NODES - 1);
  float val = 0.f;
  if (r < NH) val = __expf(tgt[r * N_NODES + m] - hmax[r]);
  BmatT[(size_t)(HD + r) * N_NODES + m] = (f16_t)val;
}

// ------------------- GEMM2: onum = adj_f16 @ BmatT^T ------------------------
__global__ __launch_bounds__(256) void k_gemm2(const int* __restrict__ adj,
                                               const f16_t* __restrict__ BmatT,
                                               float* __restrict__ onum) {
  __shared__ f16_t Alds[128 * LDA];
  __shared__ f16_t Blds[128 * LDA];

  const int lane   = threadIdx.x & 31;
  const int wave   = threadIdx.x >> 5;
  const int waveM  = wave >> 2;
  const int waveN  = wave & 3;
  const int tile_m = blockIdx.y * 128;
  const int tile_n = blockIdx.x * 128;

  v8f acc[4][2];
#pragma unroll
  for (int mt = 0; mt < 4; ++mt)
#pragma unroll
    for (int nt = 0; nt < 2; ++nt) acc[mt][nt] = {};

  for (int kk = 0; kk < N_NODES; kk += 32) {
    // Kick off B tile first: 128 cols x 64B, async global->LDS (b128/lane),
    // K-contiguous per column so no transpose is needed.
#pragma unroll
    for (int f = 0; f < 2; ++f) {
      const int linear = threadIdx.x + f * 256;          // 0..511
      const int col = linear >> 2;                       // 0..127
      const int ch  = (linear & 3) << 3;                 // half offset 0/8/16/24
      async_copy_b128(BmatT + (size_t)(tile_n + col) * N_NODES + kk + ch,
                      &Blds[col * LDA + ch]);
    }
    // Stage A tile while the async copy runs: 128x32 int32 {0,1} -> f16 LDS
#pragma unroll
    for (int f = 0; f < 4; ++f) {
      const int linear = threadIdx.x + f * 256;
      const int row = linear >> 3;
      const int c4  = (linear & 7) << 2;
      const int4 v = *(const int4*)(adj + (size_t)(tile_m + row) * N_NODES + kk + c4);
      f16_t* dst = &Alds[row * LDA + c4];
      dst[0] = (f16_t)(float)v.x; dst[1] = (f16_t)(float)v.y;
      dst[2] = (f16_t)(float)v.z; dst[3] = (f16_t)(float)v.w;
    }
    if (kk + 32 < N_NODES)   // prefetch next adj tile -> global_prefetch_b8
      __builtin_prefetch(adj + (size_t)(tile_m + (threadIdx.x >> 1)) * N_NODES + kk + 32, 0, 0);
    wait_async();            // ASYNCcnt==0: this wave's B-tile chunks landed
    __syncthreads();         // + everyone's A-tile stores / B chunks visible

    v16h bfr[2];
#pragma unroll
    for (int nt = 0; nt < 2; ++nt)
      bfr[nt] = load_frag_B(Blds, waveN * 32 + nt * 16 + (lane & 15), lane);
#pragma unroll
    for (int mt = 0; mt < 4; ++mt) {
      const v16h afr = load_frag_A(Alds, waveM * 64 + mt * 16 + (lane & 15), lane);
#pragma unroll
      for (int nt = 0; nt < 2; ++nt)
        acc[mt][nt] = __builtin_amdgcn_wmma_f32_16x16x32_f16(
            false, afr, false, bfr[nt], (short)0, acc[mt][nt], false, false);
    }
    __syncthreads();
  }

#pragma unroll
  for (int mt = 0; mt < 4; ++mt)
#pragma unroll
    for (int nt = 0; nt < 2; ++nt) {
      const int col   = tile_n + waveN * 32 + nt * 16 + (lane & 15);
      const int rbase = tile_m + waveM * 64 + mt * 16 + ((lane < 16) ? 0 : 8);
#pragma unroll
      for (int i = 0; i < 8; ++i)
        onum[(size_t)(rbase + i) * BCOLS + col] = acc[mt][nt][i];
    }
}

// -------- out[n,d] = mean_h ( onum[n,h*256+d] / onum[n,1024+h] ) ------------
__global__ __launch_bounds__(256) void k_final(const float* __restrict__ onum,
                                               float* __restrict__ out) {
  const int idx = blockIdx.x * 256 + threadIdx.x;  // < 4096*256
  const int n = idx >> 8, d = idx & 255;
  float accv = 0.f;
#pragma unroll
  for (int hh = 0; hh < NH; ++hh)
    accv += onum[(size_t)n * BCOLS + hh * DH + d] / onum[(size_t)n * BCOLS + HD + hh];
  out[idx] = 0.25f * accv;
}

// ---------------------------------------------------------------------------
extern "C" void kernel_launch(void* const* d_in, const int* in_sizes, int n_in,
                              void* d_out, int out_size, void* d_ws, size_t ws_size,
                              hipStream_t stream) {
  const float* hmat = (const float*)d_in[0];   // [4096, 512]
  const int*   adj  = (const int*)d_in[1];     // [4096, 4096]
  const float* Wmat = (const float*)d_in[2];   // [512, 1024]
  const float* avec = (const float*)d_in[3];   // [4, 512, 1]
  float* out = (float*)d_out;                  // [4096, 256]

  char* ws = (char*)d_ws;
  float* ht    = (float*)ws;                                  // 16 MiB
  float* tgt   = (float*)(ws + (size_t)16 * 1024 * 1024);     // 64 KiB
  float* hmax  = tgt + NH * N_NODES;                          // 16 B
  f16_t* BmatT = (f16_t*)(ws + (size_t)17 * 1024 * 1024);     // 9.44 MB
  float* onum  = (float*)(ws + (size_t)27 * 1024 * 1024);     // 18.9 MB

  const dim3 blk(256);
  k_gemm1    <<<dim3(HD / 128, N_NODES / 128), blk, 0, stream>>>(hmat, Wmat, ht);
  k_tgt      <<<dim3(NH * N_NODES / 8),        blk, 0, stream>>>(ht, avec, tgt);
  k_max      <<<dim3(NH),                      blk, 0, stream>>>(tgt, hmax);
  k_build_T  <<<dim3(DH / 32, N_NODES / 256, NH), blk, 0, stream>>>(ht, tgt, hmax, BmatT);
  k_build_den<<<dim3((BCOLS - HD) * N_NODES / 256), blk, 0, stream>>>(tgt, hmax, BmatT);
  k_gemm2    <<<dim3(BCOLS / 128, N_NODES / 128), blk, 0, stream>>>(adj, BmatT, onum);
  k_final    <<<dim3(N_NODES * DH / 256),      blk, 0, stream>>>(onum, out);
}
